// SkeletonGNN_10892037062762
// MI455X (gfx1250) — matrix operand order repaired and checked by reading, not compiled, not run on hardware
//
#include <hip/hip_runtime.h>
#include <hip/hip_bf16.h>

// ---------------------------------------------------------------------------
// SkeletonGNN for MI455X (gfx1250, wave32).
//  - Edge/node MLPs run on v_wmma_f32_16x16x32_f16 (16 rows per wave tile).
//  - x / aggr tables (2 MB each) are L2-resident; scatter via atomic f32 adds.
//  - LDS staging marshals gathered rows into CDNA5 WMMA fragment layouts.
// ---------------------------------------------------------------------------

typedef __attribute__((ext_vector_type(16))) _Float16 v16h;
typedef __attribute__((ext_vector_type(8)))  float    v8f;

#define DIM   5      // NODE_DIM
#define HID   64     // HIDDEN
#define KPAD  32     // padded K for first GEMM (2*DIM=10 -> 32)
#define WAVES 8      // 256-thread block = 8 wave32

__device__ __forceinline__ void lds_wave_fence() {
  // DS ops are in-order per wave; wait for outstanding LDS stores so other
  // lanes' staged data is visible, and compiler-barrier the access order.
  asm volatile("s_wait_dscnt 0" ::: "memory");
}

// Build A fragment (16x32 f16, M x K) from a row-major LDS tile.
// Contiguous K order: lanes 0-15 hold row M=lane, elements 0..7 -> K 0..7,
// 8..15 -> K 8..15; lanes 16-31 hold row M=lane-16, K 16..23 / 24..31.
__device__ __forceinline__ v16h load_a_frag(const _Float16* row, int kb0, int kb1) {
  v16h a;
#pragma unroll
  for (int j = 0; j < 8; ++j) { a[j] = row[kb0 + j]; a[8 + j] = row[kb1 + j]; }
  return a;
}

// Build B fragment (32x16 f16, K x N) from an N-major LDS weight tile
// (wcol points at [n][kbB]): element j -> K = kbB + j, N = lane%16.
__device__ __forceinline__ v16h load_b_frag(const _Float16* wcol) {
  v16h b;
#pragma unroll
  for (int j = 0; j < 16; ++j) b[j] = wcol[j];
  return b;
}

// ---------------------------------------------------------------------------
// Edge kernel: msg = relu([x_i,x_j] @ W1 + b1) @ W2 + b2 ; atomicAdd -> aggr
// ---------------------------------------------------------------------------
__global__ __launch_bounds__(256) void gnn_edge_kernel(
    const float* __restrict__ x,      // [N, DIM]
    const int*   __restrict__ ei,     // [2, E] (row0 = src, row1 = dst)
    const float* __restrict__ W1,     // [2*DIM, HID]
    const float* __restrict__ b1,     // [HID]
    const float* __restrict__ W2,     // [HID, DIM]
    const float* __restrict__ b2,     // [DIM]
    float* __restrict__ aggr,         // [N, DIM]
    int nEdges)
{
  __shared__ _Float16 sW1[HID * KPAD];        // N-major: [n][k], zero-padded K
  __shared__ _Float16 sW2[16 * HID];          // N-major: [n][k], n padded 5->16
  __shared__ float    sB1[HID];
  __shared__ float    sB2[16];
  __shared__ _Float16 sA[WAVES][16 * KPAD];   // per-wave 16x32 A staging
  __shared__ _Float16 sH[WAVES][16 * HID];    // per-wave 16x64 hidden staging
  __shared__ int      sDst[WAVES][16];

  const int tid = threadIdx.x;

  // Cooperative weight load: f32 -> f16, transposed to N-major, zero-padded.
  for (int i = tid; i < HID * KPAD; i += 256) {
    int n = i / KPAD, k = i % KPAD;
    sW1[i] = (k < 2 * DIM) ? (_Float16)W1[k * HID + n] : (_Float16)0.f;
  }
  for (int i = tid; i < 16 * HID; i += 256) {
    int n = i / HID, k = i % HID;
    sW2[i] = (n < DIM) ? (_Float16)W2[k * DIM + n] : (_Float16)0.f;
  }
  if (tid < HID) sB1[tid] = b1[tid];
  if (tid < 16)  sB2[tid] = (tid < DIM) ? b2[tid] : 0.f;
  __syncthreads();

  const int  wave  = tid >> 5;
  const int  lane  = tid & 31;
  const int  hlane = lane & 15;
  const bool hi    = lane >= 16;
  const int  tile  = blockIdx.x * WAVES + wave;
  const int  e0    = tile * 16;
  if (e0 >= nEdges) return;                    // wave-uniform exit: EXEC stays all-1s

  const int* srcIdx = ei;
  const int* dstIdx = ei + nEdges;

  // Warm the edge-index stream ahead of this block (global_prefetch_b8).
  if (lane == 0) {
    __builtin_prefetch(srcIdx + e0 + 4096, 0, 0);
    __builtin_prefetch(dstIdx + e0 + 4096, 0, 0);
  }

  // Stage gathered rows: lanes 0-15 own one edge each.
  if (!hi) {
    const int e = e0 + hlane;
    _Float16* row = &sA[wave][hlane * KPAD];
    if (e < nEdges) {
      const int s = srcIdx[e];
      const int d = dstIdx[e];
      sDst[wave][hlane] = d;
#pragma unroll
      for (int k = 0; k < DIM; ++k) {
        row[k]       = (_Float16)x[d * DIM + k];   // x_i = x[dst]
        row[DIM + k] = (_Float16)x[s * DIM + k];   // x_j = x[src]
      }
#pragma unroll
      for (int k = 2 * DIM; k < KPAD; ++k) row[k] = (_Float16)0.f;
    } else {
      sDst[wave][hlane] = -1;
#pragma unroll
      for (int k = 0; k < KPAD; ++k) row[k] = (_Float16)0.f;
    }
  }
  lds_wave_fence();

  const int kb0 = hi ? 16 : 0;
  const int kb1 = hi ? 24 : 8;
  const int kbB = hi ? 16 : 0;

  const v16h afrag = load_a_frag(&sA[wave][hlane * KPAD], kb0, kb1);

  // Stage 1: hidden = relu(A @ W1 + b1); four 16-wide N tiles.
#pragma unroll
  for (int t = 0; t < 4; ++t) {
    const int n0 = t * 16;
    const v16h bfrag = load_b_frag(&sW1[(n0 + hlane) * KPAD + kbB]);
    v8f c = {};
    c = __builtin_amdgcn_wmma_f32_16x16x32_f16(false, afrag, false, bfrag,
                                               (short)0, c, false, false);
    const float bias = sB1[n0 + hlane];
    _Float16* hcol = &sH[wave][n0 + hlane];   // column n0+hlane, stride HID
#pragma unroll
    for (int r = 0; r < 8; ++r) {
      float v = c[r] + bias;
      v = v > 0.f ? v : 0.f;
      const int m = hi ? (r + 8) : r;         // D layout: hi half holds M=8..15
      hcol[m * HID] = (_Float16)v;
    }
  }
  lds_wave_fence();

  // Stage 2: msg = H @ W2 + b2; K=64 as two chained 16x16x32 WMMAs.
  v8f d = {};
#pragma unroll
  for (int kc = 0; kc < 2; ++kc) {
    const int ko = kc * 32;
    const v16h a2 = load_a_frag(&sH[wave][hlane * HID + ko], kb0, kb1);
    const v16h b2f = load_b_frag(&sW2[hlane * HID + ko + kbB]);
    d = __builtin_amdgcn_wmma_f32_16x16x32_f16(false, a2, false, b2f,
                                               (short)0, d, false, false);
  }

  // Scatter-add (segment_sum): columns 0..4 are valid message features.
  if (hlane < DIM) {
    const float bias = sB2[hlane];
#pragma unroll
    for (int r = 0; r < 8; ++r) {
      const int m = hi ? (r + 8) : r;
      const int dnode = sDst[wave][m];
      if (dnode >= 0) atomicAdd(&aggr[dnode * DIM + hlane], d[r] + bias);
    }
  }
}

// ---------------------------------------------------------------------------
// Node kernel: x = relu([x,aggr] @ U1 + bu1) @ U2 + bu2 + x   (residual)
// ---------------------------------------------------------------------------
__global__ __launch_bounds__(256) void gnn_node_kernel(
    float*       __restrict__ x,      // [N, DIM] updated in place
    const float* __restrict__ aggr,   // [N, DIM]
    const float* __restrict__ U1,     // [2*DIM, HID]
    const float* __restrict__ bu1,    // [HID]
    const float* __restrict__ U2,     // [HID, DIM]
    const float* __restrict__ bu2,    // [DIM]
    int nNodes)
{
  __shared__ _Float16 sU1[HID * KPAD];
  __shared__ _Float16 sU2[16 * HID];
  __shared__ float    sB1[HID];
  __shared__ float    sB2[16];
  __shared__ _Float16 sA[WAVES][16 * KPAD];
  __shared__ _Float16 sH[WAVES][16 * HID];

  const int tid = threadIdx.x;
  for (int i = tid; i < HID * KPAD; i += 256) {
    int n = i / KPAD, k = i % KPAD;
    sU1[i] = (k < 2 * DIM) ? (_Float16)U1[k * HID + n] : (_Float16)0.f;
  }
  for (int i = tid; i < 16 * HID; i += 256) {
    int n = i / HID, k = i % HID;
    sU2[i] = (n < DIM) ? (_Float16)U2[k * DIM + n] : (_Float16)0.f;
  }
  if (tid < HID) sB1[tid] = bu1[tid];
  if (tid < 16)  sB2[tid] = (tid < DIM) ? bu2[tid] : 0.f;
  __syncthreads();

  const int  wave  = tid >> 5;
  const int  lane  = tid & 31;
  const int  hlane = lane & 15;
  const bool hi    = lane >= 16;
  const int  tile  = blockIdx.x * WAVES + wave;
  const int  i0    = tile * 16;
  if (i0 >= nNodes) return;                    // wave-uniform exit

  if (!hi) {
    const int i = i0 + hlane;
    _Float16* row = &sA[wave][hlane * KPAD];
    if (i < nNodes) {
#pragma unroll
      for (int k = 0; k < DIM; ++k) {
        row[k]       = (_Float16)x[i * DIM + k];
        row[DIM + k] = (_Float16)aggr[i * DIM + k];
      }
#pragma unroll
      for (int k = 2 * DIM; k < KPAD; ++k) row[k] = (_Float16)0.f;
    } else {
#pragma unroll
      for (int k = 0; k < KPAD; ++k) row[k] = (_Float16)0.f;
    }
  }
  lds_wave_fence();

  const int kb0 = hi ? 16 : 0;
  const int kb1 = hi ? 24 : 8;
  const int kbB = hi ? 16 : 0;

  const v16h afrag = load_a_frag(&sA[wave][hlane * KPAD], kb0, kb1);

#pragma unroll
  for (int t = 0; t < 4; ++t) {
    const int n0 = t * 16;
    const v16h bfrag = load_b_frag(&sU1[(n0 + hlane) * KPAD + kbB]);
    v8f c = {};
    c = __builtin_amdgcn_wmma_f32_16x16x32_f16(false, afrag, false, bfrag,
                                               (short)0, c, false, false);
    const float bias = sB1[n0 + hlane];
    _Float16* hcol = &sH[wave][n0 + hlane];
#pragma unroll
    for (int r = 0; r < 8; ++r) {
      float v = c[r] + bias;
      v = v > 0.f ? v : 0.f;
      const int m = hi ? (r + 8) : r;
      hcol[m * HID] = (_Float16)v;
    }
  }
  lds_wave_fence();

  v8f d = {};
#pragma unroll
  for (int kc = 0; kc < 2; ++kc) {
    const int ko = kc * 32;
    const v16h a2 = load_a_frag(&sH[wave][hlane * HID + ko], kb0, kb1);
    const v16h b2f = load_b_frag(&sU2[hlane * HID + ko + kbB]);
    d = __builtin_amdgcn_wmma_f32_16x16x32_f16(false, a2, false, b2f,
                                               (short)0, d, false, false);
  }

  if (hlane < DIM) {
    const float bias = sB2[hlane];
#pragma unroll
    for (int r = 0; r < 8; ++r) {
      const int m = hi ? (r + 8) : r;
      const int i = i0 + m;
      if (i < nNodes) x[i * DIM + hlane] += d[r] + bias;   // residual update
    }
  }
}

// ---------------------------------------------------------------------------
// Readout: out = x @ Wr + br  (5x5 — scalar VALU, 25 FMA/node)
// ---------------------------------------------------------------------------
__global__ __launch_bounds__(256) void gnn_readout_kernel(
    const float* __restrict__ x, const float* __restrict__ Wr,
    const float* __restrict__ br, float* __restrict__ out, int nNodes)
{
  const int i = blockIdx.x * blockDim.x + threadIdx.x;
  if (i >= nNodes) return;
  float xi[DIM];
#pragma unroll
  for (int k = 0; k < DIM; ++k) xi[k] = x[i * DIM + k];
#pragma unroll
  for (int n = 0; n < DIM; ++n) {
    float a = br[n];
#pragma unroll
    for (int k = 0; k < DIM; ++k) a = fmaf(xi[k], Wr[k * DIM + n], a);
    out[i * DIM + n] = a;
  }
}

// ---------------------------------------------------------------------------
extern "C" void kernel_launch(void* const* d_in, const int* in_sizes, int n_in,
                              void* d_out, int out_size, void* d_ws, size_t ws_size,
                              hipStream_t stream) {
  (void)n_in; (void)out_size; (void)ws_size;

  const float* x_in = (const float*)d_in[0];
  const int*   ei   = (const int*)  d_in[1];
  const float* W1   = (const float*)d_in[2];
  const float* b1   = (const float*)d_in[3];
  const float* W2   = (const float*)d_in[4];
  const float* b2   = (const float*)d_in[5];
  const float* U1   = (const float*)d_in[6];
  const float* bu1  = (const float*)d_in[7];
  const float* U2   = (const float*)d_in[8];
  const float* bu2  = (const float*)d_in[9];
  const float* Wr   = (const float*)d_in[10];
  const float* br   = (const float*)d_in[11];

  const int N = in_sizes[0] / DIM;       // 100000
  const int E = in_sizes[1] / 2;         // 1600000
  const int L = in_sizes[3] > 0 ? in_sizes[3] / HID : 3;  // N_LAYERS from b1

  float* xcur = (float*)d_ws;            // [N, DIM]
  float* aggr = xcur + (size_t)N * DIM;  // [N, DIM]

  hipMemcpyAsync(xcur, x_in, (size_t)N * DIM * sizeof(float),
                 hipMemcpyDeviceToDevice, stream);

  const int tilesE  = (E + 15) / 16;
  const int blocksE = (tilesE + WAVES - 1) / WAVES;
  const int tilesN  = (N + 15) / 16;
  const int blocksN = (tilesN + WAVES - 1) / WAVES;

  for (int l = 0; l < L; ++l) {
    hipMemsetAsync(aggr, 0, (size_t)N * DIM * sizeof(float), stream);
    gnn_edge_kernel<<<blocksE, 256, 0, stream>>>(
        xcur, ei,
        W1 + (size_t)l * 2 * DIM * HID, b1 + (size_t)l * HID,
        W2 + (size_t)l * HID * DIM,     b2 + (size_t)l * DIM,
        aggr, E);
    gnn_node_kernel<<<blocksN, 256, 0, stream>>>(
        xcur, aggr,
        U1 + (size_t)l * 2 * DIM * HID, bu1 + (size_t)l * HID,
        U2 + (size_t)l * HID * DIM,     bu2 + (size_t)l * DIM,
        N);
  }

  gnn_readout_kernel<<<(N + 255) / 256, 256, 0, stream>>>(
      xcur, Wr, br, (float*)d_out, N);
}
